// Self_Attention_49383533970025
// MI455X (gfx1250) — matrix operand order repaired and checked by reading, not compiled, not run on hardware
//
#include <hip/hip_runtime.h>
#include <hip/hip_bf16.h>
#include <math.h>

// ---------------------------------------------------------------------------
// Self-attention (B=4, T=4096, C=1024) for gfx1250 (MI455X, CDNA5, wave32).
//
// Roofline: ~412 GFLOP total vs ~145 MB mandatory HBM traffic -> AI ~2900 F/B,
// far past the ~430 F/B crossover for a bf16-WMMA ceiling at 23.3 TB/s HBM.
// => compute-bound: everything runs through v_wmma_f32_16x16x32_bf16, and the
// 256 MB score tensor is never materialized (flash-attention streaming
// softmax; writing it out would add ~1 GB of traffic, ~45 us, doubling time).
//
// v2 changes vs v1 (reasoned, no runtime feedback):
//  - softmax parallelized across all 8 waves (P fragments built in-register,
//    alpha/l broadcast by __shfl) -> no idle waves, one less barrier/step
//  - 64-key blocks (half the barrier/zero/atomic overhead per key)
//  - LDS S tile padded to 65 f32/row -> conflict-free per-row reads
//  - fused QKV projection with x-tile staged once in LDS (8x less A traffic),
//    same for the O-tile in the output projection
//  - global_prefetch of the next K / Vt key block
// ---------------------------------------------------------------------------

typedef __bf16 bf16_t;
typedef bf16_t v4bf  __attribute__((ext_vector_type(4)));
typedef bf16_t v8bf  __attribute__((ext_vector_type(8)));
typedef bf16_t v16bf __attribute__((ext_vector_type(16)));
typedef float  v8f   __attribute__((ext_vector_type(8)));

#define B_  4
#define T_  4096
#define C_  1024
#define BT_ (B_ * T_)                       // 16384 rows
#define LDP 1032                            // padded LDS row (bf16 elems)
static const size_t CC_  = (size_t)C_ * C_;    // 1,048,576
static const size_t BTC_ = (size_t)BT_ * C_;   // 16,777,216

__device__ inline v8f wmma_bf16(v16bf a, v16bf b, v8f c) {
  // D(16x16 f32) = A(16x32 bf16) * B(32x16 bf16) + C
  return __builtin_amdgcn_wmma_f32_16x16x32_bf16(
      false, a, false, b, (short)0, c, false, false);
}

// Compose a 16-element bf16 fragment from two 8-element (16B) loads.
__device__ inline v16bf make_frag(const bf16_t* p0, const bf16_t* p1) {
  v8bf lo = *(const v8bf*)p0;
  v8bf hi = *(const v8bf*)p1;
  v16bf f;
#pragma unroll
  for (int i = 0; i < 8; ++i) { f[i] = lo[i]; f[i + 8] = hi[i]; }
  return f;
}

// ---------------------------------------------------------------------------
// Kernel 1: weights fp32 -> bf16.
// ---------------------------------------------------------------------------
__global__ __launch_bounds__(256) void cvt_w_kernel(
    const float* __restrict__ Wq, const float* __restrict__ Wk,
    const float* __restrict__ Wv, const float* __restrict__ Wo,
    bf16_t* __restrict__ wbf) {
  size_t i = (size_t)blockIdx.x * blockDim.x + threadIdx.x;
  if (i < CC_) {
    wbf[i]           = (bf16_t)Wq[i];
    wbf[CC_ + i]     = (bf16_t)Wk[i];
    wbf[2 * CC_ + i] = (bf16_t)Wv[i];
    wbf[3 * CC_ + i] = (bf16_t)Wo[i];
  }
}

// ---------------------------------------------------------------------------
// Kernel 2: fused Q/K/V projection.  q[m,n] = sum_k x[m,k] * W[n,k]
// Block = 8 waves; block tile = 16 rows x 128 cols; the 16x1024 x-tile is
// converted to bf16 and staged ONCE in LDS, then shared by all 8 waves and
// all three GEMMs (3 B-fragments + 3 WMMAs per A-fragment).
// V is stored transposed (Vt[b][d][t]) so P*V B-fragments are contiguous;
// the Vt store itself is one contiguous 16B store per lane.
// ---------------------------------------------------------------------------
__global__ __launch_bounds__(256) void proj_qkv_kernel(
    const float* __restrict__ x, const bf16_t* __restrict__ wbf,
    bf16_t* __restrict__ Qb, bf16_t* __restrict__ Kb,
    bf16_t* __restrict__ Vtb) {
  __shared__ __align__(16) bf16_t xs[16][LDP];

  const int wave = threadIdx.x >> 5;
  const int lane = threadIdx.x & 31;
  const int n15  = lane & 15;
  const int half = lane >> 4;
  const int m0 = blockIdx.x * 16;
  const int n0 = blockIdx.y * 128 + wave * 16;

  // Stage x tile (16 x 1024 fp32) as bf16 in LDS, float4 at a time.
  for (int i = threadIdx.x; i < 4096; i += 256) {
    const int idx = i * 4;
    const int r = idx >> 10, c = idx & 1023;
    const float4 v = *(const float4*)(x + (size_t)(m0 + r) * C_ + c);
    v4bf pk;
    pk[0] = (bf16_t)v.x; pk[1] = (bf16_t)v.y;
    pk[2] = (bf16_t)v.z; pk[3] = (bf16_t)v.w;
    *(v4bf*)&xs[r][c] = pk;
  }
  __syncthreads();

  const bf16_t* wq = wbf + (size_t)(n0 + n15) * C_;
  const bf16_t* wk = wq + CC_;
  const bf16_t* wv = wk + CC_;

  v8f aq = {}, ak = {}, av = {};
  for (int k0 = 0; k0 < C_; k0 += 32) {
    const bf16_t* pa = &xs[n15][k0 + half * 8];
    v16bf a = make_frag(pa, pa + 16);
    const int kb = k0 + half * 8;
    aq = wmma_bf16(a, make_frag(wq + kb, wq + kb + 16), aq);
    ak = wmma_bf16(a, make_frag(wk + kb, wk + kb + 16), ak);
    av = wmma_bf16(a, make_frag(wv + kb, wv + kb + 16), av);
  }

  // Q, K row-major.
#pragma unroll
  for (int r = 0; r < 8; ++r) {
    const size_t off = (size_t)(m0 + r + 8 * half) * C_ + n0 + n15;
    Qb[off] = (bf16_t)aq[r];
    Kb[off] = (bf16_t)ak[r];
  }
  // V transposed: D rows r+8*half are contiguous in t -> one 16B store.
  {
    const int bb = m0 / T_;
    const int t0 = m0 - bb * T_;
    v8bf st;
#pragma unroll
    for (int r = 0; r < 8; ++r) st[r] = (bf16_t)av[r];
    *(v8bf*)(Vtb + ((size_t)bb * C_ + n0 + n15) * T_ + t0 + 8 * half) = st;
  }
}

// ---------------------------------------------------------------------------
// Kernel 3: flash attention (causal).  Block = 16 query rows, 8 waves; wave w
// owns d-slice [w*128, w*128+128) => O accumulator = 8 x v8f (64 VGPRs/lane).
// Per 64-key block:
//  (1) zero LDS S (16 x 64, rows padded to 65 f32 -> bank-conflict-free)
//  (2) each wave adds its d-slice partial of S (16 WMMAs + ds_add_f32)
//  (3) EVERY wave runs the online softmax for query row n15 (lane-local),
//      builds its two P A-fragments in registers, broadcasts alpha via shfl
//  (4) O rescale + P*V (16 WMMAs, Vt B-fragments contiguous)
// ---------------------------------------------------------------------------
__global__ __launch_bounds__(256) void attn_kernel(
    const bf16_t* __restrict__ Qb, const bf16_t* __restrict__ Kb,
    const bf16_t* __restrict__ Vtb, bf16_t* __restrict__ Ob) {
  __shared__ float Sld[16 * 65];

  const int lane = threadIdx.x & 31;
  const int wave = threadIdx.x >> 5;
  const int n15  = lane & 15;
  const int half = lane >> 4;
  const int bb  = blockIdx.y;
  const int q0  = blockIdx.x * 16;
  const int dw0 = wave * 128;
  const size_t rowQ = (size_t)bb * T_ + q0;
  const float scale = 0.03125f;            // 1/sqrt(1024)
  const int qg = q0 + n15;                 // this lane's query row (global s)

  // Preload Q A-fragments for the d-slice.
  v16bf qf[4];
  {
    const bf16_t* qrow = Qb + (rowQ + n15) * (size_t)C_ + dw0;
#pragma unroll
    for (int kk = 0; kk < 4; ++kk) {
      const bf16_t* p = qrow + kk * 32 + half * 8;
      qf[kk] = make_frag(p, p + 16);
    }
  }

  v8f o[8];
#pragma unroll
  for (int t = 0; t < 8; ++t) o[t] = (v8f){};
  float mrow = -3.0e38f, lrow = 0.0f;      // stats for query row n15

  for (int s0 = 0; s0 < q0 + 16; s0 += 64) {
    // Prefetch next key block (speculative, gfx1250 global_prefetch).
    if (s0 + 64 < q0 + 16) {
      const bf16_t* kn = Kb + ((size_t)bb * T_ + s0 + 64 + lane) * C_ + dw0;
      __builtin_prefetch(kn, 0, 2);
      __builtin_prefetch(kn + (size_t)32 * C_, 0, 2);
#pragma unroll
      for (int X = 0; X < 4; ++X)
        __builtin_prefetch(
            Vtb + ((size_t)bb * C_ + dw0 + lane + 32 * X) * T_ + s0 + 64, 0, 2);
    }

    // (1) zero shared S
    for (int i = threadIdx.x; i < 16 * 65; i += 256) Sld[i] = 0.0f;
    __syncthreads();

    // (2) partial S over this wave's d-slice, 4 key n-tiles
#pragma unroll
    for (int j = 0; j < 4; ++j) {
      v8f s = (v8f){};
      const bf16_t* krow =
          Kb + ((size_t)bb * T_ + s0 + j * 16 + n15) * C_ + dw0;
#pragma unroll
      for (int kk = 0; kk < 4; ++kk) {
        const bf16_t* p = krow + kk * 32 + half * 8;
        s = wmma_bf16(qf[kk], make_frag(p, p + 16), s);
      }
#pragma unroll
      for (int r = 0; r < 8; ++r)
        atomicAdd(&Sld[(r + 8 * half) * 65 + j * 16 + n15], s[r]);
    }
    __syncthreads();

    // (3) online softmax for query row n15 (all lanes, all waves)
    const float* srow = &Sld[n15 * 65];
    float mnew = mrow;
#pragma unroll
    for (int c = 0; c < 64; ++c) {
      float v = (s0 + c <= qg) ? srow[c] * scale : -3.0e38f;
      mnew = fmaxf(mnew, v);
    }
    const float alpha = __expf(mrow - mnew);
    float sum = 0.0f;
#pragma unroll
    for (int c = 0; c < 64; ++c) {
      float v = (s0 + c <= qg) ? srow[c] * scale : -3.0e38f;
      sum += __expf(v - mnew);
    }
    // Build the two P A-fragments (cols f*32 + {half*8+e, 16+half*8+e}).
    v16bf pf[2];
#pragma unroll
    for (int f = 0; f < 2; ++f) {
#pragma unroll
      for (int e = 0; e < 8; ++e) {
        const int c0 = f * 32 + half * 8 + e;
        const int c1 = c0 + 16;
        float v0 = (s0 + c0 <= qg) ? srow[c0] * scale : -3.0e38f;
        float v1 = (s0 + c1 <= qg) ? srow[c1] * scale : -3.0e38f;
        pf[f][e]     = (bf16_t)__expf(v0 - mnew);
        pf[f][e + 8] = (bf16_t)__expf(v1 - mnew);
      }
    }
    lrow = lrow * alpha + sum;
    mrow = mnew;

    // (4) rescale O (alpha of row r+8*half broadcast via shfl), then P*V
#pragma unroll
    for (int r = 0; r < 8; ++r) {
      const float al = __shfl(alpha, r + 8 * half);
#pragma unroll
      for (int t = 0; t < 8; ++t) o[t][r] *= al;
    }
#pragma unroll
    for (int f = 0; f < 2; ++f) {
#pragma unroll
      for (int t = 0; t < 8; ++t) {
        const bf16_t* vtc = Vtb +
            ((size_t)bb * C_ + dw0 + t * 16 + n15) * T_ + s0 + f * 32 +
            half * 8;
        o[t] = wmma_bf16(pf[f], make_frag(vtc, vtc + 16), o[t]);
      }
    }
    __syncthreads();   // S reads done before next iteration's zeroing
  }

  // Epilogue: divide by l (broadcast per D-row) and store bf16 O.
#pragma unroll
  for (int r = 0; r < 8; ++r) {
    const float rl = 1.0f / __shfl(lrow, r + 8 * half);
#pragma unroll
    for (int t = 0; t < 8; ++t)
      Ob[(rowQ + r + 8 * half) * (size_t)C_ + dw0 + t * 16 + n15] =
          (bf16_t)(o[t][r] * rl);
  }
}

// ---------------------------------------------------------------------------
// Kernel 4: output projection  out[m,n] = sum_k O[m,k] * Wo[n,k] + bo[n]
// O-tile staged once in LDS (shared across the 8 waves).
// ---------------------------------------------------------------------------
__global__ __launch_bounds__(256) void out_proj_kernel(
    const bf16_t* __restrict__ Ob, const bf16_t* __restrict__ Wob,
    const float* __restrict__ bo, float* __restrict__ out) {
  __shared__ __align__(16) bf16_t xs[16][LDP];

  const int wave = threadIdx.x >> 5;
  const int lane = threadIdx.x & 31;
  const int n15  = lane & 15;
  const int half = lane >> 4;
  const int m0 = blockIdx.x * 16;
  const int n0 = blockIdx.y * 128 + wave * 16;

  for (int i = threadIdx.x; i < 2048; i += 256) {
    const int idx = i * 8;
    const int r = idx >> 10, c = idx & 1023;
    *(v8bf*)&xs[r][c] = *(const v8bf*)(Ob + (size_t)(m0 + r) * C_ + c);
  }
  __syncthreads();

  const bf16_t* wrow = Wob + (size_t)(n0 + n15) * C_;
  v8f acc = {};
  for (int k0 = 0; k0 < C_; k0 += 32) {
    const bf16_t* pa = &xs[n15][k0 + half * 8];
    v16bf a = make_frag(pa, pa + 16);
    const bf16_t* pb = wrow + k0 + half * 8;
    acc = wmma_bf16(a, make_frag(pb, pb + 16), acc);
  }

  const float bias = bo[n0 + n15];
#pragma unroll
  for (int r = 0; r < 8; ++r)
    out[(size_t)(m0 + r + 8 * half) * C_ + n0 + n15] = acc[r] + bias;
}

// ---------------------------------------------------------------------------
// Launcher.  Workspace (bf16 elems): 4*C*C weights | Q | K | Vt | O = 136 MB.
// ---------------------------------------------------------------------------
extern "C" void kernel_launch(void* const* d_in, const int* in_sizes, int n_in,
                              void* d_out, int out_size, void* d_ws,
                              size_t ws_size, hipStream_t stream) {
  (void)in_sizes; (void)n_in; (void)out_size; (void)ws_size;
  const float* x  = (const float*)d_in[0];
  const float* Wq = (const float*)d_in[1];
  const float* Wk = (const float*)d_in[2];
  const float* Wv = (const float*)d_in[3];
  const float* Wo = (const float*)d_in[4];
  const float* bo = (const float*)d_in[5];

  bf16_t* ws  = (bf16_t*)d_ws;
  bf16_t* Wbf = ws;                     // 4 * C*C
  bf16_t* Qb  = ws + 4 * CC_;
  bf16_t* Kb  = Qb + BTC_;
  bf16_t* Vtb = Kb + BTC_;
  bf16_t* Obf = Vtb + BTC_;

  cvt_w_kernel<<<(unsigned)((CC_ + 255) / 256), 256, 0, stream>>>(
      Wq, Wk, Wv, Wo, Wbf);

  proj_qkv_kernel<<<dim3(BT_ / 16, C_ / 128), 256, 0, stream>>>(
      x, Wbf, Qb, Kb, Vtb);

  attn_kernel<<<dim3(T_ / 16, B_), 256, 0, stream>>>(Qb, Kb, Vtb, Obf);

  out_proj_kernel<<<dim3(BT_ / 16, C_ / 128), 256, 0, stream>>>(
      Obf, Wbf + 3 * CC_, bo, (float*)d_out);
}
